// CNNTransformer_40080634806914
// MI455X (gfx1250) — matrix operand order repaired and checked by reading, not compiled, not run on hardware
//
#include <hip/hip_runtime.h>

// ---------------------------------------------------------------------------
// CDNA5 (gfx1250) CNN-Transformer forward.  All heavy GEMM-like work goes
// through v_wmma_f32_16x16x32_bf16 (wave32, fp32 accumulate).  Weight tiles
// are staged global->LDS with GLOBAL_LOAD_ASYNC_TO_LDS_B128 when available.
// ---------------------------------------------------------------------------

typedef __bf16 bf16;
typedef __attribute__((ext_vector_type(16))) __bf16 v16bf;
typedef __attribute__((ext_vector_type(8)))  float  v8f;

union FragU {
  v16bf v;
  bf16  h[16];
  uint4 u[2];
};

__device__ __forceinline__ v8f wmma_bf16(v16bf a, v16bf b, v8f c) {
  // D = A(16x32) * B(32x16) + C, fp32 accum
  return __builtin_amdgcn_wmma_f32_16x16x32_bf16(false, a, false, b, (short)0, c,
                                                 false, false);
}

__device__ __forceinline__ float gelu_exact(float x) {
  return 0.5f * x * (1.0f + erff(x * 0.70710678118654752f));
}

__device__ __forceinline__ float sel8(const float (&a)[8], int j) {
  float r = a[0];
#pragma unroll
  for (int t = 1; t < 8; ++t) r = (j == t) ? a[t] : r;
  return r;
}

__device__ __forceinline__ unsigned int pack2_bf16(float a, float b) {
  union { bf16 h; unsigned short u; } ua, ub;
  ua.h = (bf16)a;
  ub.h = (bf16)b;
  return ((unsigned int)ub.u << 16) | (unsigned int)ua.u;
}

// --- optional CDNA5 async global->LDS copy (ASYNCcnt path) -----------------
#if __has_builtin(__builtin_amdgcn_global_load_async_to_lds_b128)
#define HAVE_ASYNC_LDS 1
typedef int v4i_ __attribute__((vector_size(16)));
__device__ __forceinline__ void async_copy_b128(const void* g, void* l) {
  __builtin_amdgcn_global_load_async_to_lds_b128(
      (__attribute__((address_space(1))) v4i_*)(v4i_*)(void*)g,
      (__attribute__((address_space(3))) v4i_*)(v4i_*)l, 0, 0);
}
__device__ __forceinline__ void async_wait0() {
  asm volatile("s_wait_asynccnt 0" ::: "memory");
}
#endif

// ---------------------------------------------------------------------------
// fp32 -> bf16 weight conversion
// ---------------------------------------------------------------------------
__global__ __launch_bounds__(256) void k_cvt_bf16(const float* __restrict__ s,
                                                  bf16* __restrict__ d, int n) {
  int i = blockIdx.x * 256 + threadIdx.x;
  if (i < n) d[i] = (bf16)s[i];
}

// ---------------------------------------------------------------------------
// Embedding conv1: 3->256, k3 s2 p1, fused (x + pos) and exact GELU. Small
// K (27), done on VALU.
// ---------------------------------------------------------------------------
__global__ __launch_bounds__(256) void k_conv1(const float* __restrict__ x,
                                               const float* __restrict__ pos,
                                               const float* __restrict__ w,
                                               const float* __restrict__ bias,
                                               float* __restrict__ out) {
  int idx = blockIdx.x * 256 + threadIdx.x;  // (b, co, oy, ox), 112x112 out
  int ox = idx % 112;
  int t = idx / 112;
  int oy = t % 112;
  t /= 112;
  int co = t % 256;
  int b = t / 256;
  float acc = bias[co];
#pragma unroll
  for (int ci = 0; ci < 3; ++ci) {
#pragma unroll
    for (int r = 0; r < 3; ++r) {
      int iy = oy * 2 - 1 + r;
      if (iy < 0 || iy >= 224) continue;
#pragma unroll
      for (int s = 0; s < 3; ++s) {
        int ix = ox * 2 - 1 + s;
        if (ix < 0 || ix >= 224) continue;
        float v = x[(((size_t)b * 3 + ci) * 224 + iy) * 224 + ix] +
                  pos[iy * 224 + ix];
        acc += v * w[((co * 3 + ci) * 3 + r) * 3 + s];
      }
    }
  }
  out[idx] = gelu_exact(acc);
}

// ---------------------------------------------------------------------------
// Generic implicit-GEMM conv, bf16 WMMA.  M=Cout, N=B*Hout*Wout, K=Cin*k*k
// (K is a multiple of 32 for all shapes here).  Block = 256 threads (8 waves),
// tile = 64(M) x 64(N), K-chunks of 32.  Each wave owns a 32x16 sub-tile
// (2 WMMAs/chunk, shared B fragment).  A tile is a straight bf16 copy
// (async global->LDS when available); B tile is an im2col gather with
// hoisted n-decode and carry-incremented k-decode (no divisions in the
// inner loop).  act: 0=none, 1=exact GELU.  resid may be null.
// ---------------------------------------------------------------------------
__global__ __launch_bounds__(256) void k_conv_wmma(
    const float* __restrict__ in, const bf16* __restrict__ w,
    const float* __restrict__ bias, const float* __restrict__ resid,
    float* __restrict__ out, int Bn, int Cin, int Hin, int Win, int Cout,
    int Hout, int Wout, int ksz, int stride, int pad, int act) {
  __shared__ __align__(16) bf16 aT[64][32];  // [m][k]
  __shared__ __align__(16) bf16 bT[64][32];  // [n][k] (transposed for B frags)

  const int tid = threadIdx.x;
  const int lane = tid & 31;
  const int wv = tid >> 5;
  const int tileM = (wv >> 2) * 32;  // 0 or 32
  const int tileN = (wv & 3) * 16;   // 0..48
  const int HWo = Hout * Wout;
  const int Ntot = Bn * HWo;
  const int kk2 = ksz * ksz;
  const int Ktot = Cin * kk2;
  const int m0 = blockIdx.y * 64;
  const int n0 = blockIdx.x * 64;

  // per-thread staging assignment: one row, 8 consecutive k
  const int trow = tid >> 2;         // 0..63 (row in A and B tiles)
  const int kbase = (tid & 3) * 8;   // 0, 8, 16, 24

  // A: weight row
  const int grow = m0 + trow;
  const bool arow_ok = grow < Cout;
  const bf16* wrow = w + (size_t)grow * Ktot;

  // B: hoisted n-decode (one division chain per thread, outside the K loop)
  const int gn_th = n0 + trow;
  const bool nok = gn_th < Ntot;
  int iy0 = 0, ix0 = 0;
  const float* inb = in;
  if (nok) {
    int bb = gn_th / HWo;
    int rem = gn_th - bb * HWo;
    int oy = rem / Wout, ox = rem - oy * Wout;
    iy0 = oy * stride - pad;
    ix0 = ox * stride - pad;
    inb = in + (size_t)bb * Cin * Hin * Win;
  }

  v8f acc0 = {}, acc1 = {};

  for (int kc = 0; kc < Ktot; kc += 32) {
    // --- stage A (weights): 64x32 bf16, one b128 per thread ---
    if (arow_ok) {
#ifdef HAVE_ASYNC_LDS
      async_copy_b128(&wrow[kc + kbase], &aT[trow][kbase]);
#else
      *(uint4*)&aT[trow][kbase] = *(const uint4*)&wrow[kc + kbase];
#endif
    } else {
      uint4 z = {};
      *(uint4*)&aT[trow][kbase] = z;
    }

    // --- stage B (im2col gather): 32(K) x 64(N), stored [n][k] ---
    {
      int gk = kc + kbase;
      int ci = gk / kk2;
      int rs = gk - ci * kk2;
      int r = rs / ksz;
      int s = rs - r * ksz;
      float v[8];
#pragma unroll
      for (int j = 0; j < 8; ++j) {
        float x = 0.f;
        if (nok) {
          int iy = iy0 + r, ix = ix0 + s;
          if (iy >= 0 && iy < Hin && ix >= 0 && ix < Win)
            x = inb[((size_t)ci * Hin + iy) * Win + ix];
        }
        v[j] = x;
        // carry-increment (s, r, ci)
        if (++s == ksz) {
          s = 0;
          if (++r == ksz) {
            r = 0;
            ++ci;
          }
        }
      }
      uint4 bv;
      bv.x = pack2_bf16(v[0], v[1]);
      bv.y = pack2_bf16(v[2], v[3]);
      bv.z = pack2_bf16(v[4], v[5]);
      bv.w = pack2_bf16(v[6], v[7]);
      *(uint4*)&bT[trow][kbase] = bv;
    }
#ifdef HAVE_ASYNC_LDS
    async_wait0();
#endif
    __syncthreads();

    // --- fragments per documented CDNA5 bf16 WMMA layouts ---
    FragU fa0, fa1, fb;
    int arow0 = tileM + (lane & 15);
    int klo = (lane >> 4) * 8;  // lanes 16-31 hold K=8..15 / 24..31
    fa0.u[0] = *(const uint4*)&aT[arow0][klo];
    fa0.u[1] = *(const uint4*)&aT[arow0][16 + klo];
    fa1.u[0] = *(const uint4*)&aT[arow0 + 16][klo];
    fa1.u[1] = *(const uint4*)&aT[arow0 + 16][16 + klo];
    int bn = tileN + (lane & 15);
    int koff = (lane >> 4) * 8;  // lanes 16-31 hold K=8..15 / 24..31
    const uint4* bp = (const uint4*)&bT[bn][koff];
    fb.u[0] = bp[0];
    fb.u[1] = bp[2];
    acc0 = wmma_bf16(fa0.v, fb.v, acc0);
    acc1 = wmma_bf16(fa1.v, fb.v, acc1);
    __syncthreads();
  }

  // --- epilogue: bias + activation + optional residual ---
  int gn = n0 + tileN + (lane & 15);
  if (gn < Ntot) {
    int bb = gn / HWo;
    int rem = gn - bb * HWo;
#pragma unroll
    for (int j = 0; j < 8; ++j) {
      int co0 = m0 + tileM + ((lane >> 4) * 8) + j;
      if (co0 < Cout) {
        float v = acc0[j] + bias[co0];
        if (act == 1) v = gelu_exact(v);
        size_t oidx = ((size_t)bb * Cout + co0) * HWo + rem;
        if (resid) v += resid[oidx];
        out[oidx] = v;
      }
      int co1 = co0 + 16;
      if (co1 < Cout) {
        float v = acc1[j] + bias[co1];
        if (act == 1) v = gelu_exact(v);
        size_t oidx = ((size_t)bb * Cout + co1) * HWo + rem;
        if (resid) v += resid[oidx];
        out[oidx] = v;
      }
    }
  }
}

// ---------------------------------------------------------------------------
// AdaptiveAvgPool2d 56 -> 14 (4x4 mean)
// ---------------------------------------------------------------------------
__global__ __launch_bounds__(256) void k_pool4(const float* __restrict__ in,
                                               float* __restrict__ out) {
  int idx = blockIdx.x * 256 + threadIdx.x;  // (b, c, oy, ox), 14x14
  int ox = idx % 14;
  int t = idx / 14;
  int oy = t % 14;
  t /= 14;  // t = b*256 + c
  const float* p = in + ((size_t)t * 56 + oy * 4) * 56 + ox * 4;
  float s = 0.f;
#pragma unroll
  for (int dy = 0; dy < 4; ++dy)
#pragma unroll
    for (int dx = 0; dx < 4; ++dx) s += p[dy * 56 + dx];
  out[idx] = s * (1.0f / 16.0f);
}

// ---------------------------------------------------------------------------
// Linear 196 -> 784 over M = B*256 = 2048 rows, bf16 WMMA.
// out[m*784 + n] = sum_k in[m*196+k] * wb[n*196+k] + bias[n]
// ---------------------------------------------------------------------------
__global__ __launch_bounds__(256) void k_linear_wmma(
    const float* __restrict__ in, const bf16* __restrict__ wb,
    const float* __restrict__ bias, float* __restrict__ out) {
  __shared__ __align__(16) bf16 aT[32][32];
  __shared__ __align__(16) bf16 bT[64][32];
  const int tid = threadIdx.x;
  const int lane = tid & 31;
  const int wv = tid >> 5;
  const int tileM = (wv >> 2) * 16;
  const int tileN = (wv & 3) * 16;
  const int m0 = blockIdx.y * 32;
  const int n0 = blockIdx.x * 64;
  v8f acc = {};

  for (int kc = 0; kc < 196; kc += 32) {
#pragma unroll
    for (int j = 0; j < 4; ++j) {
      int li = tid * 4 + j;
      int m = li >> 5, k = li & 31;
      int gk = kc + k;
      aT[m][k] = (gk < 196) ? (bf16)in[(size_t)(m0 + m) * 196 + gk] : (bf16)0.f;
    }
#pragma unroll
    for (int j = 0; j < 8; ++j) {
      int li = tid * 8 + j;
      int n = li >> 5, k = li & 31;
      int gn = n0 + n, gk = kc + k;
      bT[n][k] = (gn < 784 && gk < 196) ? (bf16)wb[(size_t)gn * 196 + gk]
                                        : (bf16)0.f;
    }
    __syncthreads();
    FragU fa, fb;
    int arow = tileM + (lane & 15);
    int klo = (lane >> 4) * 8;
    fa.u[0] = *(const uint4*)&aT[arow][klo];
    fa.u[1] = *(const uint4*)&aT[arow][16 + klo];
    int bn = tileN + (lane & 15);
    int koff = (lane >> 4) * 8;
    const uint4* bp = (const uint4*)&bT[bn][koff];
    fb.u[0] = bp[0];
    fb.u[1] = bp[2];
    acc = wmma_bf16(fa.v, fb.v, acc);
    __syncthreads();
  }

  int gn = n0 + tileN + (lane & 15);
  if (gn < 784) {
#pragma unroll
    for (int j = 0; j < 8; ++j) {
      int m = m0 + tileM + ((lane >> 4) * 8) + j;
      out[(size_t)m * 784 + gn] = acc[j] + bias[gn];
    }
  }
}

// ---------------------------------------------------------------------------
// FeatureNorm: LayerNorm over 256 channels at each (b, n) location, bias-free.
// One wave per location; lane handles 8 channels.
// ---------------------------------------------------------------------------
__global__ __launch_bounds__(256) void k_fnorm(const float* __restrict__ x,
                                               const float* __restrict__ w,
                                               float* __restrict__ out) {
  int wv = threadIdx.x >> 5, lane = threadIdx.x & 31;
  int gid = blockIdx.x * 8 + wv;  // 0 .. 8*784-1
  int b = gid / 784, n = gid % 784;
  const float* xp = x + (size_t)b * 256 * 784 + n;
  float v[8], s = 0.f, sq = 0.f;
#pragma unroll
  for (int i = 0; i < 8; ++i) {
    v[i] = xp[(size_t)(lane + 32 * i) * 784];
    s += v[i];
    sq += v[i] * v[i];
  }
#pragma unroll
  for (int off = 1; off < 32; off <<= 1) {
    s += __shfl_xor(s, off, 32);
    sq += __shfl_xor(sq, off, 32);
  }
  float mean = s * (1.0f / 256.0f);
  float var = sq * (1.0f / 256.0f) - mean * mean;
  float inv = rsqrtf(var + 1e-5f);
  float* op = out + (size_t)b * 256 * 784 + n;
#pragma unroll
  for (int i = 0; i < 8; ++i) {
    int c = lane + 32 * i;
    op[(size_t)c * 784] = (v[i] - mean) * inv * w[c];
  }
}

// ---------------------------------------------------------------------------
// Flash attention, bf16 WMMA.  qkv layout: (head, batch, 96, 784),
// ch 0-31 = Q, 32-63 = K, 64-95 = V.  hd = 32, N = 784 (49 query tiles of 16).
// One wave per (b, h, qtile).  y layout: (B, 256, 784).
// Per 32-key chunk: S = Q K^T (2 WMMAs, K-dim = hd = 32), online softmax,
// P staged through per-wave LDS tile, then O^T += V^T P^T (2 WMMAs).
// ---------------------------------------------------------------------------
__global__ __launch_bounds__(256) void k_attn(const float* __restrict__ qkv,
                                              float* __restrict__ y) {
  __shared__ __align__(16) bf16 p_lds[8][16][32];  // [wave][query][key]

  int wv = threadIdx.x >> 5, lane = threadIdx.x & 31;
  int gid = blockIdx.x * 8 + wv;  // 0..3135
  int qt = gid % 49;
  int t = gid / 49;
  int b = t & 7;
  int h = t >> 3;
  const float* qp = qkv + (size_t)(h * 8 + b) * 96 * 784;
  const float* kp = qp + 32 * 784;
  const float* vp = qp + 64 * 784;

  const int q15 = lane & 15;
  const int half = lane >> 4;  // 0 or 1
  const int klo = half * 8;
  const float scale = 0.17677669529663687f;  // 1/sqrt(32)

  // Q A-fragment (16 queries x 32 dims), scale folded into bf16 conversion
  FragU qa;
  int qrow = qt * 16 + q15;
#pragma unroll
  for (int e = 0; e < 16; ++e) {
    int d = (e < 8) ? (klo + e) : (16 + klo + (e - 8));
    qa.h[e] = (bf16)(qp[(size_t)d * 784 + qrow] * scale);
  }

  v8f O0 = {}, O1 = {};  // O^T: rows d0-15 / d16-31, cols = queries
  float rmax[8], rsum[8];
#pragma unroll
  for (int j = 0; j < 8; ++j) {
    rmax[j] = -3e38f;
    rsum[j] = 0.f;
  }

  for (int kc = 0; kc < 800; kc += 32) {  // keys padded to 800 with masking
    int key0 = kc + q15;
    int key1 = kc + 16 + q15;
    bool ok0 = key0 < 784, ok1 = key1 < 784;

    // K B-fragments: B[k=d][n=key]
    FragU kb0, kb1;
#pragma unroll
    for (int e = 0; e < 16; ++e) {
      int d = (e < 8) ? (klo + e) : (16 + klo + (e - 8));
      kb0.h[e] = ok0 ? (bf16)kp[(size_t)d * 784 + key0] : (bf16)0.f;
      kb1.h[e] = ok1 ? (bf16)kp[(size_t)d * 784 + key1] : (bf16)0.f;
    }
    v8f z0 = {}, z1 = {};
    v8f S0 = wmma_bf16(qa.v, kb0.v, z0);
    v8f S1 = wmma_bf16(qa.v, kb1.v, z1);

    // online softmax across this wave-half's 16 key lanes
    float alpha[8], p0v[8], p1v[8];
#pragma unroll
    for (int j = 0; j < 8; ++j) {
      float s0 = ok0 ? S0[j] : -3e38f;
      float s1 = ok1 ? S1[j] : -3e38f;
      float m = fmaxf(s0, s1);
#pragma unroll
      for (int off = 1; off < 16; off <<= 1)
        m = fmaxf(m, __shfl_xor(m, off, 16));
      float nm = fmaxf(rmax[j], m);
      float al = __expf(rmax[j] - nm);
      float p0 = ok0 ? __expf(s0 - nm) : 0.f;
      float p1 = ok1 ? __expf(s1 - nm) : 0.f;
      float rs = p0 + p1;
#pragma unroll
      for (int off = 1; off < 16; off <<= 1) rs += __shfl_xor(rs, off, 16);
      rsum[j] = rsum[j] * al + rs;
      rmax[j] = nm;
      alpha[j] = al;
      p0v[j] = p0;
      p1v[j] = p1;
    }

    // stage P (16 queries x 32 keys) through LDS for the transpose
    int rbase = half * 8;
#pragma unroll
    for (int j = 0; j < 8; ++j) {
      p_lds[wv][rbase + j][q15] = (bf16)p0v[j];
      p_lds[wv][rbase + j][16 + q15] = (bf16)p1v[j];
    }
    __builtin_amdgcn_wave_barrier();
    asm volatile("" ::: "memory");

    // P^T B-fragment: lane n = query, k = key
    FragU pb;
#pragma unroll
    for (int e = 0; e < 16; ++e) pb.h[e] = p_lds[wv][q15][(e < 8) ? (klo + e) : (16 + klo + (e - 8))];

    // V^T A-fragments: rows = dims, K = keys
    FragU va0, va1;
#pragma unroll
    for (int e = 0; e < 16; ++e) {
      int kl = (e < 8) ? (klo + e) : (16 + klo + (e - 8));
      int key = kc + kl;
      bool ok = key < 784;
      va0.h[e] = ok ? (bf16)vp[(size_t)q15 * 784 + key] : (bf16)0.f;
      va1.h[e] = ok ? (bf16)vp[(size_t)(16 + q15) * 784 + key] : (bf16)0.f;
    }

    // per-query rescale of running O (alpha lives in S-layout halves)
    float own = sel8(alpha, q15 & 7);
    float oth = __shfl_xor(own, 16, 32);
    float myalpha = (((q15 >> 3) != 0) == (half != 0)) ? own : oth;
#pragma unroll
    for (int j = 0; j < 8; ++j) {
      O0[j] *= myalpha;
      O1[j] *= myalpha;
    }
    O0 = wmma_bf16(va0.v, pb.v, O0);
    O1 = wmma_bf16(va1.v, pb.v, O1);
    __builtin_amdgcn_wave_barrier();
    asm volatile("" ::: "memory");
  }

  float owns = sel8(rsum, q15 & 7);
  float oths = __shfl_xor(owns, 16, 32);
  float myrsum = (((q15 >> 3) != 0) == (half != 0)) ? owns : oths;
  float inv = 1.0f / myrsum;
  int qg = qt * 16 + q15;
  size_t ybase = ((size_t)b * 256 + h * 32) * 784 + qg;
#pragma unroll
  for (int j = 0; j < 8; ++j) {
    int d = half * 8 + j;
    y[ybase + (size_t)d * 784] = O0[j] * inv;
    y[ybase + (size_t)(16 + d) * 784] = O1[j] * inv;
  }
}

// ---------------------------------------------------------------------------
// Final LayerNorm over the (28,28) spatial plane per (b, c); affine over plane.
// One wave per (b, c) row of 784.
// ---------------------------------------------------------------------------
__global__ __launch_bounds__(256) void k_lnf(const float* __restrict__ x,
                                             const float* __restrict__ w,
                                             const float* __restrict__ bias,
                                             float* __restrict__ out) {
  int wv = threadIdx.x >> 5, lane = threadIdx.x & 31;
  int gid = blockIdx.x * 8 + wv;  // 0..2047 = b*256 + c
  const float* xp = x + (size_t)gid * 784;
  float s = 0.f, sq = 0.f;
  for (int i = 0; i < 25; ++i) {
    int idx = lane + 32 * i;
    if (idx < 784) {
      float v = xp[idx];
      s += v;
      sq += v * v;
    }
  }
#pragma unroll
  for (int off = 1; off < 32; off <<= 1) {
    s += __shfl_xor(s, off, 32);
    sq += __shfl_xor(sq, off, 32);
  }
  float mean = s * (1.0f / 784.0f);
  float var = sq * (1.0f / 784.0f) - mean * mean;
  float inv = rsqrtf(var + 1e-5f);
  float* op = out + (size_t)gid * 784;
  for (int i = 0; i < 25; ++i) {
    int idx = lane + 32 * i;
    if (idx < 784) op[idx] = (xp[idx] - mean) * inv * w[idx] + bias[idx];
  }
}

// ---------------------------------------------------------------------------
// Host-side orchestration
// ---------------------------------------------------------------------------
extern "C" void kernel_launch(void* const* d_in, const int* in_sizes, int n_in,
                              void* d_out, int out_size, void* d_ws,
                              size_t ws_size, hipStream_t stream) {
  (void)in_sizes;
  (void)n_in;
  (void)out_size;
  (void)ws_size;  // requires ~141 MB of workspace

  const float* X    = (const float*)d_in[0];
  const float* POS  = (const float*)d_in[1];
  const float* EW1  = (const float*)d_in[2];
  const float* EB1  = (const float*)d_in[3];
  const float* EW2  = (const float*)d_in[4];
  const float* EB2  = (const float*)d_in[5];
  const float* LW   = (const float*)d_in[6];
  const float* LB   = (const float*)d_in[7];
  const float* LNFW = (const float*)d_in[8];
  const float* LNFB = (const float*)d_in[9];

  char* ws = (char*)d_ws;
  // main regions
  const size_t H1_OFF = 0;                      // 102,760,448 B (dead after conv2)
  const size_t H2_OFF = 102760448;              // 25,690,112 B
  const size_t POOL_OFF = 128450560;            //  1,605,632 B
  const size_t O_OFF = 130056192;               //  6,422,528 B
  const size_t WSMALL_OFF = 136478720;          //  ~4 MB (ew2 / lw bf16)
  // overlays inside the h1 region (valid once conv2 has consumed h1)
  const size_t T_OFF = 0;                       // 6.5 MB
  const size_t QKV_OFF = (size_t)8 << 20;       // 19.3 MB
  const size_t ATT_OFF = (size_t)28 << 20;      // 6.5 MB
  const size_t MLPH_OFF = (size_t)36 << 20;     // 25.7 MB
  const size_t WBF_OFF = (size_t)64 << 20;      // up to 6 MB of bf16 weights

  float* h1   = (float*)(ws + H1_OFF);
  float* h2   = (float*)(ws + H2_OFF);
  float* pool = (float*)(ws + POOL_OFF);
  float* o    = (float*)(ws + O_OFF);
  bf16* wsml  = (bf16*)(ws + WSMALL_OFF);
  float* tbuf = (float*)(ws + T_OFF);
  float* qkv  = (float*)(ws + QKV_OFF);
  float* att  = (float*)(ws + ATT_OFF);
  float* mlph = (float*)(ws + MLPH_OFF);
  bf16* wbf   = (bf16*)(ws + WBF_OFF);

  auto cvt = [&](const float* src, bf16* dst, int n) {
    k_cvt_bf16<<<(n + 255) / 256, 256, 0, stream>>>(src, dst, n);
  };
  auto conv = [&](const float* in, const float* w32, int wsz, const float* bias,
                  const float* resid, float* outp, int Bn, int Cin, int Hin,
                  int Win, int Cout, int Hout, int Wout, int ksz, int stride,
                  int pad, int act, bf16* wdst) {
    cvt(w32, wdst, wsz);
    int Ntot = Bn * Hout * Wout;
    dim3 g((Ntot + 63) / 64, (Cout + 63) / 64);
    k_conv_wmma<<<g, 256, 0, stream>>>(in, wdst, bias, resid, outp, Bn, Cin,
                                       Hin, Win, Cout, Hout, Wout, ksz, stride,
                                       pad, act);
  };

  // --- embedding ---
  k_conv1<<<(8 * 256 * 112 * 112) / 256, 256, 0, stream>>>(X, POS, EW1, EB1, h1);
  conv(h1, EW2, 256 * 256 * 9, EB2, nullptr, h2, 8, 256, 112, 112, 256, 56, 56,
       3, 2, 1, /*gelu*/ 1, wsml);
  k_pool4<<<(8 * 256 * 196) / 256, 256, 0, stream>>>(h2, pool);
  cvt(LW, wsml, 784 * 196);
  k_linear_wmma<<<dim3(13, 64), 256, 0, stream>>>(pool, wsml, LB, o);

  // --- transformer blocks ---
  for (int blk = 0; blk < 3; ++blk) {
    int base = 10 + blk * 24;
    const float* LN1 = (const float*)d_in[base + 0];
    const float* LN2 = (const float*)d_in[base + 1];
    const float* PW  = (const float*)d_in[base + 18];
    const float* PB  = (const float*)d_in[base + 19];
    const float* MW1 = (const float*)d_in[base + 20];
    const float* MB1 = (const float*)d_in[base + 21];
    const float* MW2 = (const float*)d_in[base + 22];
    const float* MB2 = (const float*)d_in[base + 23];

    // attention branch
    k_fnorm<<<784, 256, 0, stream>>>(o, LN1, tbuf);
    for (int i = 0; i < 8; ++i) {
      const float* AW = (const float*)d_in[base + 2 + i];
      const float* AB = (const float*)d_in[base + 10 + i];
      int ks = (i % 6) * 2 + 1;
      int pd = i % 6;
      conv(tbuf, AW, 96 * 256 * ks * ks, AB, nullptr,
           qkv + (size_t)i * 8 * 96 * 784, 8, 256, 28, 28, 96, 28, 28, ks, 1,
           pd, 0, wbf);
    }
    k_attn<<<392, 256, 0, stream>>>(qkv, att);
    conv(att, PW, 256 * 256, PB, /*resid*/ o, o, 8, 256, 28, 28, 256, 28, 28, 1,
         1, 0, 0, wbf);

    // MLP branch
    k_fnorm<<<784, 256, 0, stream>>>(o, LN2, tbuf);
    conv(tbuf, MW1, 1024 * 256 * 9, MB1, nullptr, mlph, 8, 256, 28, 28, 1024,
         28, 28, 3, 1, 1, /*gelu*/ 1, wbf);
    conv(mlph, MW2, 256 * 1024 * 9, MB2, /*resid*/ o, o, 8, 1024, 28, 28, 256,
         28, 28, 3, 1, 1, 0, wbf);
  }

  // --- final LayerNorm over (28,28) ---
  k_lnf<<<256, 256, 0, stream>>>(o, LNFW, LNFB, (float*)d_out);
}
